// AdditiveAttention_5712306504431
// MI455X (gfx1250) — matrix operand order repaired and checked by reading, not compiled, not run on hardware
//
#include <hip/hip_runtime.h>
#include <math.h>
#include <stdint.h>

// Problem constants (from reference)
#define BB   2
#define SS   1024
#define SQ   512
#define KVH  256
#define QH   256
#define AH   128

typedef __attribute__((ext_vector_type(16))) _Float16 v16h;
typedef __attribute__((ext_vector_type(8)))  float    v8f;

// Fast tanh: gfx1250 has V_TANH_F32 (TRANS unit). Guarded so compile never breaks.
#if __has_builtin(__builtin_amdgcn_tanhf)
#define FAST_TANH(x) __builtin_amdgcn_tanhf(x)
#elif __has_builtin(__builtin_amdgcn_tanh_f32)
#define FAST_TANH(x) __builtin_amdgcn_tanh_f32(x)
#else
#define FAST_TANH(x) tanhf(x)
#endif

// Async memory->LDS copy (gfx1250): per-lane 16B copy, tracked by ASYNCcnt.
// LDS destination = low 32 bits of generic shared pointer (LDS aperture maps
// addr[31:0] directly to the LDS offset).
__device__ __forceinline__ void async_copy_b128(const float* gsrc, float* lds_dst)
{
    const unsigned lds_off = (unsigned)(uintptr_t)lds_dst;
    const unsigned long long gaddr = (unsigned long long)(uintptr_t)gsrc;
    asm volatile("global_load_async_to_lds_b128 %0, %1, off"
                 :: "v"(lds_off), "v"(gaddr) : "memory");
}

__device__ __forceinline__ void async_wait_all()
{
    asm volatile("s_wait_asynccnt 0x0" ::: "memory");
}

// ---------------------------------------------------------------------------
// LDS-staged WMMA GEMM: C[b] = A[b] (MxK row-major) @ B[b] (KxN row-major) + bias
// f32 in memory -> f16 operands, f32 accumulate via v_wmma_f32_16x16x32_f16.
// Block = 128 threads (4 waves). Each block: 16x64 C tile; each wave: 16x16.
// A tile (16x32) staged once per K-step, shared by all 4 waves. Tiles staged
// with GLOBAL_LOAD_ASYNC_TO_LDS_B128 (ASYNCcnt) -- no VGPR round trip.
// grid = (N/64, M/16, nbatch). Requires N%64==0, K%32==0, M%16==0.
// ---------------------------------------------------------------------------
#define APITCH 36   // A tile rows padded to 36 floats (144B = 9*16, keeps 16B align)
#define BPITCH 68   // B tile rows padded to 68 floats (272B = 17*16)

__global__ __launch_bounds__(128)
void wmma_gemm_f32f16(const float* __restrict__ A, const float* __restrict__ B,
                      const float* __restrict__ bias, float* __restrict__ C,
                      int M, int N, int K,
                      long long strideA, long long strideB, long long strideC)
{
    __shared__ float As[16][APITCH];
    __shared__ float Bs[32][BPITCH];

    const int tid   = threadIdx.x;
    const int lane  = tid & 31;
    const int wave  = tid >> 5;
    const int tileN = blockIdx.x * 64;
    const int tileM = blockIdx.y * 16;
    const int batch = blockIdx.z;

    const float* Ab = A + (long long)batch * strideA;
    const float* Bb = B + (long long)batch * strideB;
    float*       Cb = C + (long long)batch * strideC;

    const int half = lane >> 4;   // K-half selector per ISA 16-bit A/B layouts
    const int nrow = lane & 15;   // A row within tile / B+C column within tile

    // staging coordinates
    const int a_mr = tid >> 3;          // 0..15
    const int a_kc = (tid & 7) * 4;     // 0,4,...,28

    v8f acc = {};

#pragma unroll 2
    for (int k0 = 0; k0 < K; k0 += 32) {
        // ---- stage A tile: 16x32 f32, one async 16B copy per thread ----
        async_copy_b128(Ab + (long long)(tileM + a_mr) * K + k0 + a_kc,
                        &As[a_mr][a_kc]);
        // ---- stage B tile: 32x64 f32, four async 16B copies per thread ----
#pragma unroll
        for (int j = 0; j < 4; ++j) {
            const int idx = tid + 128 * j;     // 0..511
            const int kr  = idx >> 4;          // 0..31
            const int nc  = (idx & 15) * 4;    // 0,4,...,60
            async_copy_b128(Bb + (long long)(k0 + kr) * N + tileN + nc,
                            &Bs[kr][nc]);
        }
        async_wait_all();     // this wave's async copies landed in LDS
        __syncthreads();      // ...and everyone else's

        // ---- build WMMA operands from LDS ----
        v16h av, bv;
        // A 16x32 f16 layout: lane half -> K base +8; VGPR v<4 -> K=2v,2v+1; v>=4 -> K=16+2(v-4)+..
        const float* arow = &As[nrow][8 * half];
#pragma unroll
        for (int e = 0; e < 16; ++e) {
            const int vg = e >> 1;
            const int kk = (vg < 4 ? 2 * vg : 16 + 2 * (vg - 4)) + (e & 1);
            av[e] = (_Float16)arow[kk];
        }
        // B 32x16 f16 layout: same K order as A (lane half -> K base +8; e>=8 -> K=16+..)
#pragma unroll
        for (int e = 0; e < 16; ++e) {
            bv[e] = (_Float16)Bs[(e < 8 ? 8 * half + e : 16 + 8 * half + (e - 8))][wave * 16 + nrow];
        }

        acc = __builtin_amdgcn_wmma_f32_16x16x32_f16(
            /*neg_a=*/false, av, /*neg_b=*/false, bv,
            /*c_mod=*/(short)0, acc, /*reuse_a=*/false, /*reuse_b=*/false);

        __syncthreads();   // protect LDS tiles before next staging round
    }

    const int   ncol = tileN + wave * 16 + nrow;
    const float bval = bias ? bias[ncol] : 0.0f;

    // C layout: VGPR r holds row M = r + 8*(lane/16), column N = lane&15
#pragma unroll
    for (int r = 0; r < 8; ++r) {
        const int row = tileM + r + 8 * half;
        Cb[(long long)row * N + ncol] = acc[r] + bval;
    }
}

// ---------------------------------------------------------------------------
// score[b,q,s] = sum_a w_v[a] * tanh(pkv[b,s,a] + pq[b,q,a]) + b_v
// grid = (S/256, SQ, B), block = 256. One thread per s.
// ---------------------------------------------------------------------------
__global__ __launch_bounds__(256)
void score_kernel(const float* __restrict__ pkv, const float* __restrict__ pq,
                  const float* __restrict__ w_v, const float* __restrict__ b_v,
                  float* __restrict__ score)
{
    __shared__ float sq[AH];
    __shared__ float wv[AH];

    const int b = blockIdx.z;
    const int q = blockIdx.y;
    const int s = blockIdx.x * blockDim.x + threadIdx.x;

    if (threadIdx.x < AH) {
        sq[threadIdx.x] = pq[((long long)b * SQ + q) * AH + threadIdx.x];
        wv[threadIdx.x] = w_v[threadIdx.x];
    }
    __syncthreads();

    const float4* row = (const float4*)(pkv + ((long long)b * SS + s) * AH);
    float acc = b_v[0];
#pragma unroll 8
    for (int i = 0; i < AH / 4; ++i) {
        const float4 v = row[i];
        acc += wv[4 * i + 0] * FAST_TANH(v.x + sq[4 * i + 0]);
        acc += wv[4 * i + 1] * FAST_TANH(v.y + sq[4 * i + 1]);
        acc += wv[4 * i + 2] * FAST_TANH(v.z + sq[4 * i + 2]);
        acc += wv[4 * i + 3] * FAST_TANH(v.w + sq[4 * i + 3]);
    }
    score[((long long)b * SQ + q) * SS + s] = acc;
}

// ---------------------------------------------------------------------------
// In-place softmax over the last dim (S) of w[B*SQ, S]. One block per row.
// ---------------------------------------------------------------------------
__global__ __launch_bounds__(256)
void softmax_kernel(float* __restrict__ w)
{
    __shared__ float red[256];
    float* row = w + (long long)blockIdx.x * SS;
    const int tid = threadIdx.x;

    float mx = -INFINITY;
    for (int s = tid; s < SS; s += 256) mx = fmaxf(mx, row[s]);
    red[tid] = mx;
    __syncthreads();
    for (int off = 128; off > 0; off >>= 1) {
        if (tid < off) red[tid] = fmaxf(red[tid], red[tid + off]);
        __syncthreads();
    }
    mx = red[0];
    __syncthreads();

    float sum = 0.0f;
    for (int s = tid; s < SS; s += 256) {
        const float e = __expf(row[s] - mx);
        row[s] = e;
        sum += e;
    }
    red[tid] = sum;
    __syncthreads();
    for (int off = 128; off > 0; off >>= 1) {
        if (tid < off) red[tid] += red[tid + off];
        __syncthreads();
    }
    const float inv = 1.0f / red[0];
    __syncthreads();

    for (int s = tid; s < SS; s += 256) row[s] *= inv;
}

// ---------------------------------------------------------------------------
extern "C" void kernel_launch(void* const* d_in, const int* in_sizes, int n_in,
                              void* d_out, int out_size, void* d_ws, size_t ws_size,
                              hipStream_t stream)
{
    (void)in_sizes; (void)n_in; (void)out_size; (void)ws_size;

    const float* kv   = (const float*)d_in[0]; // (B, S, KVH)
    const float* qry  = (const float*)d_in[1]; // (B, SQ, QH)
    const float* W_kv = (const float*)d_in[2]; // (KVH, AH)
    const float* b_kv = (const float*)d_in[3]; // (AH)
    const float* W_q  = (const float*)d_in[4]; // (QH, AH)
    const float* b_q  = (const float*)d_in[5]; // (AH)
    const float* w_v  = (const float*)d_in[6]; // (AH)
    const float* b_v  = (const float*)d_in[7]; // ()

    float* out     = (float*)d_out;                            // (B, SQ, KVH)
    float* weights = out + (long long)BB * SQ * KVH;           // (B, SQ, S) — also score scratch

    float* pkv = (float*)d_ws;                                 // (B*S, AH)  = 1 MB
    float* pq  = pkv + (long long)BB * SS * AH;                // (B*SQ, AH) = 0.5 MB

    // 1) proj_kv = kv @ W_kv + b_kv   : M=B*S=2048, K=256, N=128
    {
        dim3 grid(AH / 64, (BB * SS) / 16, 1);
        wmma_gemm_f32f16<<<grid, 128, 0, stream>>>(kv, W_kv, b_kv, pkv,
                                                   BB * SS, AH, KVH, 0, 0, 0);
    }
    // 2) proj_q = q @ W_q + b_q       : M=B*SQ=1024, K=256, N=128
    {
        dim3 grid(AH / 64, (BB * SQ) / 16, 1);
        wmma_gemm_f32f16<<<grid, 128, 0, stream>>>(qry, W_q, b_q, pq,
                                                   BB * SQ, AH, QH, 0, 0, 0);
    }
    // 3) scores -> weight region of d_out
    {
        dim3 grid(SS / 256, SQ, BB);
        score_kernel<<<grid, 256, 0, stream>>>(pkv, pq, w_v, b_v, weights);
    }
    // 4) softmax in place over S
    {
        softmax_kernel<<<BB * SQ, 256, 0, stream>>>(weights);
    }
    // 5) out[b] = weights[b] (SQ x S) @ kv[b] (S x KVH)
    {
        dim3 grid(KVH / 64, SQ / 16, BB);
        wmma_gemm_f32f16<<<grid, 128, 0, stream>>>(
            weights, kv, nullptr, out,
            SQ, KVH, SS,
            (long long)SQ * SS, (long long)SS * KVH, (long long)SQ * KVH);
    }
}
